// SimGNNWithAttention_66537633349988
// MI455X (gfx1250) — matrix operand order repaired and checked by reading, not compiled, not run on hardware
//
#include <hip/hip_runtime.h>
#include <hip/hip_bf16.h>
#include <math.h>

// ---------------- problem constants (from reference) ----------------
#define NN      32768            // nodes
#define DD      128              // feature dim (D_IN == HID == 128)
#define EE      524288           // edges
#define NHEADS  4
#define CHH     32               // channels per head
#define GG      32               // graphs per batch
#define NETOT   (EE + NN)        // edges + self loops
#define BN_EPS  1e-5f
#define NSLOPE  0.2f

typedef __attribute__((ext_vector_type(16))) __bf16 v16bf;
typedef __attribute__((ext_vector_type(8)))  float  v8f;
union FragU { uint4 q[2]; v16bf v; };

__device__ __forceinline__ unsigned short f2bf(float f) {
    unsigned u = __float_as_uint(f);
    unsigned r = (u >> 16) & 1u;
    return (unsigned short)((u + 0x7FFFu + r) >> 16);   // RNE fp32->bf16
}
// order-preserving float <-> uint map for atomicMax on floats
__device__ __forceinline__ unsigned encf(float f) {
    int i = __float_as_int(f);
    return (unsigned)(i ^ ((i >> 31) | 0x80000000));
}
__device__ __forceinline__ float decf(unsigned u) {
    int i = (u & 0x80000000u) ? (int)(u ^ 0x80000000u) : (int)(~u);
    return __int_as_float(i);
}

// ---------------- fp32 -> bf16 convert ----------------
__global__ void k_convert_bf16(const float* __restrict__ x,
                               unsigned short* __restrict__ xb) {
    int t = blockIdx.x * blockDim.x + threadIdx.x;   // NN*DD threads
    xb[t] = f2bf(x[t]);
}

// ---------------- swizzle W (fp32 [128K x 128N]) into B-fragment layout ----
// layout index: (((kt*8 + nt)*32 + lane)*16 + j), lane<16: K=kt*32+j, lane>=16: K=kt*32+16+j, N=nt*16+lane%16
__global__ void k_swizzleW(const float* __restrict__ Wl,
                           const float* __restrict__ Wr,
                           unsigned short* __restrict__ out) {
    int t = blockIdx.x * blockDim.x + threadIdx.x;   // 2*128*128 threads
    int which = t >> 14;
    int r = t & 16383;
    int kt   = r >> 12;
    int nt   = (r >> 9) & 7;
    int lane = (r >> 4) & 31;
    int j    = r & 15;
    int k = kt * 32 + ((lane < 16) ? j : 16 + j);
    int n = nt * 16 + (lane & 15);
    const float* W = which ? Wr : Wl;
    out[t] = f2bf(W[k * DD + n]);
}

// ---------------- WMMA GEMM: C[M,128] = A_bf16[M,128] @ Bswz + bias -------
__global__ void __launch_bounds__(256)
k_wmma_gemm(const unsigned short* __restrict__ A,
            const unsigned short* __restrict__ Bswz,
            const float* __restrict__ bias,
            float* __restrict__ C, int M) {
    int wave = threadIdx.x >> 5;
    int lane = threadIdx.x & 31;
    int m0   = blockIdx.x * 128 + wave * 16;         // 8 waves -> 128 rows/block
    int rowA = m0 + (lane & 15);

    v8f acc[8] = {};
#pragma unroll
    for (int kt = 0; kt < 4; ++kt) {
        int kb  = kt * 32;
        int kA0 = kb + ((lane < 16) ? 0 : 16);
        int kA1 = kb + ((lane < 16) ? 8 : 24);
        FragU af;
        af.q[0] = *(const uint4*)(A + rowA * DD + kA0);
        af.q[1] = *(const uint4*)(A + rowA * DD + kA1);
#pragma unroll
        for (int nt = 0; nt < 8; ++nt) {
            FragU bfr;
            const uint4* pb = (const uint4*)(Bswz + (((kt * 8 + nt) * 32 + lane) << 4));
            bfr.q[0] = pb[0];
            bfr.q[1] = pb[1];
            acc[nt] = __builtin_amdgcn_wmma_f32_16x16x32_bf16(
                false, af.v, false, bfr.v, (short)0, acc[nt], false, false);
        }
    }
    int mrow = m0 + ((lane < 16) ? 0 : 8);
#pragma unroll
    for (int nt = 0; nt < 8; ++nt) {
        int col = nt * 16 + (lane & 15);
        float b = bias[col];
#pragma unroll
        for (int r = 0; r < 8; ++r)
            C[(mrow + r) * DD + col] = acc[nt][r] + b;
    }
}

// ---------------- per-layer init: OUT<-bias, MEN<-0, DEN<-0, bn<-0 --------
__global__ void k_init_layer(float* __restrict__ OUTB, const float* __restrict__ bias,
                             unsigned* __restrict__ MEN, float* __restrict__ DEN,
                             float* __restrict__ bns, float* __restrict__ bnq) {
    int t = blockIdx.x * blockDim.x + threadIdx.x;   // NN*DD threads
    OUTB[t] = bias[t & (DD - 1)];
    if (t < NN * NHEADS) { MEN[t] = 0u; DEN[t] = 0.f; }
    if (t < DD) { bns[t] = 0.f; bnq[t] = 0.f; }
}

// ---------------- edge pass 1: logits + segment max (wave per edge) -------
__global__ void k_edge1(const float* __restrict__ XL, const float* __restrict__ XR,
                        const int* __restrict__ srcA, const int* __restrict__ dstA,
                        const float* __restrict__ att,
                        float* __restrict__ LG, unsigned* __restrict__ MEN) {
    int e = blockIdx.x * 8 + (threadIdx.x >> 5);
    int lane = threadIdx.x & 31;
    int s, d;
    if (e < EE) { s = srcA[e]; d = dstA[e]; } else { s = d = e - EE; }
    float p[NHEADS];
#pragma unroll
    for (int i = 0; i < NHEADS; ++i) {
        int c = i * CHH + lane;
        float v = XL[s * DD + c] + XR[d * DD + c];
        v = (v > 0.f) ? v : NSLOPE * v;               // leaky_relu
        p[i] = v * att[c];
    }
#pragma unroll
    for (int off = 16; off > 0; off >>= 1)
#pragma unroll
        for (int i = 0; i < NHEADS; ++i)
            p[i] += __shfl_xor(p[i], off, 32);
    if (lane < NHEADS) {
        LG[e * NHEADS + lane] = p[lane];
        atomicMax(&MEN[d * NHEADS + lane], encf(p[lane]));
    }
}

// ---------------- edge pass 2: ex = exp(logit - m); den += ex -------------
__global__ void k_edge2(const int* __restrict__ dstA,
                        float* __restrict__ LG, const unsigned* __restrict__ MEN,
                        float* __restrict__ DEN) {
    int t = blockIdx.x * blockDim.x + threadIdx.x;    // NETOT*NHEADS threads
    int e = t >> 2, h = t & 3;
    int d = (e < EE) ? dstA[e] : (e - EE);
    float m  = decf(MEN[d * NHEADS + h]);
    float ex = __expf(LG[t] - m);
    LG[t] = ex;
    atomicAdd(&DEN[d * NHEADS + h], ex);
}

// ---------------- edge pass 3: out[dst] += xl[src] * alpha ----------------
__global__ void k_edge3(const float* __restrict__ XL,
                        const int* __restrict__ srcA, const int* __restrict__ dstA,
                        const float* __restrict__ LG, const float* __restrict__ DEN,
                        float* __restrict__ OUTB) {
    int e = blockIdx.x * 8 + (threadIdx.x >> 5);
    int lane = threadIdx.x & 31;
    int s, d;
    if (e < EE) { s = srcA[e]; d = dstA[e]; } else { s = d = e - EE; }
#pragma unroll
    for (int i = 0; i < NHEADS; ++i) {
        float alpha = LG[e * NHEADS + i] / (DEN[d * NHEADS + i] + 1e-16f);
        int c = i * CHH + lane;
        atomicAdd(&OUTB[d * DD + c], XL[s * DD + c] * alpha);
    }
}

// ---------------- BatchNorm ----------------------------------------------
__global__ void k_bn_stats(const float* __restrict__ X,
                           float* __restrict__ s, float* __restrict__ q) {
    int c = threadIdx.x;                               // blockDim = 128
    float sum = 0.f, sq = 0.f;
    for (int r = blockIdx.x; r < NN; r += gridDim.x) {
        float v = X[r * DD + c];
        sum += v; sq += v * v;
    }
    atomicAdd(&s[c], sum);
    atomicAdd(&q[c], sq);
}
__global__ void k_bn_fin(const float* __restrict__ s, const float* __restrict__ q,
                         const float* __restrict__ gamma, const float* __restrict__ beta,
                         float* __restrict__ scale, float* __restrict__ shift) {
    int c = threadIdx.x;
    float mu  = s[c] / (float)NN;
    float var = q[c] / (float)NN - mu * mu;
    float sc  = gamma[c] * rsqrtf(var + BN_EPS);
    scale[c] = sc;
    shift[c] = beta[c] - mu * sc;
}
__global__ void k_bn_apply(const float* __restrict__ X,
                           const float* __restrict__ scale, const float* __restrict__ shift,
                           float* __restrict__ Y, unsigned short* __restrict__ Yb) {
    int t = blockIdx.x * blockDim.x + threadIdx.x;     // NN*DD threads
    int c = t & (DD - 1);
    float y = fmaxf(scale[c] * X[t] + shift[c], 0.f);  // BN + ReLU
    Y[t] = y;
    Yb[t] = f2bf(y);
}

// ---------------- pooling -------------------------------------------------
__global__ void k_pool_init(float* __restrict__ ps, unsigned* __restrict__ pm,
                            float* __restrict__ pc) {
    int t = blockIdx.x * blockDim.x + threadIdx.x;     // GG*DD threads
    ps[t] = 0.f; pm[t] = 0u;
    if (t < GG) pc[t] = 0.f;
}
__global__ void k_pool_acc(const float* __restrict__ X, const int* __restrict__ batch,
                           float* __restrict__ ps, unsigned* __restrict__ pm,
                           float* __restrict__ pc) {
    int t = blockIdx.x * blockDim.x + threadIdx.x;     // NN*DD threads
    int n = t >> 7, c = t & (DD - 1);
    int b = batch[n];
    float v = X[t];
    atomicAdd(&ps[b * DD + c], v);
    atomicMax(&pm[b * DD + c], encf(v));
    if (c == 0) atomicAdd(&pc[b], 1.f);
}
__global__ void k_pool_fin(const float* __restrict__ ps, const unsigned* __restrict__ pm,
                           const float* __restrict__ pc, float* __restrict__ Gout) {
    int t = blockIdx.x * blockDim.x + threadIdx.x;     // GG*DD threads
    int b = t >> 7;
    float ct = pc[b];
    float mean = ps[t] / fmaxf(ct, 1.f);
    float mx   = (ct > 0.f) ? decf(pm[t]) : 0.f;
    Gout[t] = mean + mx;
}

// ---------------- head: cross-attn (degenerate) + MLP + sigmoid -----------
__global__ void __launch_bounds__(256)
k_head(const float* __restrict__ G1, const float* __restrict__ G2,
       const float* __restrict__ Wv, const float* __restrict__ bv,
       const float* __restrict__ Wo, const float* __restrict__ bo,
       const float* __restrict__ W1, const float* __restrict__ b1,
       const float* __restrict__ W2, const float* __restrict__ b2,
       const float* __restrict__ W3, const float* __restrict__ b3,
       float* __restrict__ out) {
    __shared__ float sA[GG * DD];    // scratch / h1
    __shared__ float sB[GG * DD];    // g1a
    __shared__ float sC[GG * DD];    // g2a
    __shared__ float sD[GG * 64];    // h2
    int tid = threadIdx.x;

    // v1 = G2 @ Wv + bv
    for (int idx = tid; idx < GG * DD; idx += blockDim.x) {
        int g = idx >> 7, n = idx & 127;
        float a = bv[n];
        for (int k = 0; k < DD; ++k) a += G2[g * DD + k] * Wv[k * DD + n];
        sA[idx] = a;
    }
    __syncthreads();
    // g1a = v1 @ Wo + bo
    for (int idx = tid; idx < GG * DD; idx += blockDim.x) {
        int g = idx >> 7, n = idx & 127;
        float a = bo[n];
        for (int k = 0; k < DD; ++k) a += sA[g * DD + k] * Wo[k * DD + n];
        sB[idx] = a;
    }
    __syncthreads();
    // v2 = G1 @ Wv + bv
    for (int idx = tid; idx < GG * DD; idx += blockDim.x) {
        int g = idx >> 7, n = idx & 127;
        float a = bv[n];
        for (int k = 0; k < DD; ++k) a += G1[g * DD + k] * Wv[k * DD + n];
        sA[idx] = a;
    }
    __syncthreads();
    // g2a = v2 @ Wo + bo
    for (int idx = tid; idx < GG * DD; idx += blockDim.x) {
        int g = idx >> 7, n = idx & 127;
        float a = bo[n];
        for (int k = 0; k < DD; ++k) a += sA[g * DD + k] * Wo[k * DD + n];
        sC[idx] = a;
    }
    __syncthreads();
    // h1 = relu(comb @ W1 + b1), comb = [g1a, g2a, g1a*g2a, |g1a-g2a|]  (32 x 512)
    for (int idx = tid; idx < GG * DD; idx += blockDim.x) {
        int g = idx >> 7, n = idx & 127;
        float a = b1[n];
        for (int k = 0; k < 512; ++k) {
            int kk = k & 127;
            float x1v = sB[g * DD + kk], x2v = sC[g * DD + kk];
            float cb = (k < 128) ? x1v
                     : (k < 256) ? x2v
                     : (k < 384) ? x1v * x2v
                                 : fabsf(x1v - x2v);
            a += cb * W1[k * DD + n];
        }
        sA[idx] = fmaxf(a, 0.f);
    }
    __syncthreads();
    // h2 = relu(h1 @ W2 + b2)   (32 x 64)
    for (int idx = tid; idx < GG * 64; idx += blockDim.x) {
        int g = idx >> 6, n = idx & 63;
        float a = b2[n];
        for (int k = 0; k < DD; ++k) a += sA[g * DD + k] * W2[k * 64 + n];
        sD[idx] = fmaxf(a, 0.f);
    }
    __syncthreads();
    // out = sigmoid(h2 @ W3 + b3)  (32 x 1)
    if (tid < GG) {
        float a = b3[0];
        for (int k = 0; k < 64; ++k) a += sD[tid * 64 + k] * W3[k];
        out[tid] = 1.f / (1.f + __expf(-a));
    }
}

// =====================================================================
extern "C" void kernel_launch(void* const* d_in, const int* in_sizes, int n_in,
                              void* d_out, int out_size, void* d_ws, size_t ws_size,
                              hipStream_t stream) {
    // Input order = setup_inputs() dict insertion order:
    // 0:x1 1:x2 | gat_i (i=0..2) base 2+8i: Wl bl Wr br att bias gamma beta
    // 26..33: mha Wq bq Wk bk Wv bv Wo bo | 34..39: mlp W1 b1 W2 b2 W3 b3
    // 40:edge_index1 41:edge_index2 42:batch1 43:batch2
    const float* x[2]  = { (const float*)d_in[0], (const float*)d_in[1] };
    const int*   ei[2] = { (const int*)d_in[40], (const int*)d_in[41] };
    const int*   bt[2] = { (const int*)d_in[42], (const int*)d_in[43] };

    // workspace layout
    char* w = (char*)d_ws;
    size_t o = 0;
    float*          XF   = (float*)(w + o);          o += (size_t)NN * DD * 4;   // 16 MB
    unsigned short* XB   = (unsigned short*)(w + o); o += (size_t)NN * DD * 2;   //  8 MB
    float*          XL   = (float*)(w + o);          o += (size_t)NN * DD * 4;
    float*          XR   = (float*)(w + o);          o += (size_t)NN * DD * 4;
    float*          OUTB = (float*)(w + o);          o += (size_t)NN * DD * 4;
    float*          LG   = (float*)(w + o);          o += (size_t)NETOT * NHEADS * 4;
    unsigned*       MEN  = (unsigned*)(w + o);       o += (size_t)NN * NHEADS * 4;
    float*          DEN  = (float*)(w + o);          o += (size_t)NN * NHEADS * 4;
    unsigned short* WSWZ = (unsigned short*)(w + o); o += (size_t)2 * DD * DD * 2;
    float*          BNS  = (float*)(w + o);          o += DD * 4;
    float*          BNQ  = (float*)(w + o);          o += DD * 4;
    float*          BSC  = (float*)(w + o);          o += DD * 4;
    float*          BSH  = (float*)(w + o);          o += DD * 4;
    float*          PS   = (float*)(w + o);          o += (size_t)GG * DD * 4;
    unsigned*       PM   = (unsigned*)(w + o);       o += (size_t)GG * DD * 4;
    float*          PC   = (float*)(w + o);          o += GG * 4;
    float*          GB[2];
    GB[0] = (float*)(w + o);                         o += (size_t)GG * DD * 4;
    GB[1] = (float*)(w + o);                         o += (size_t)GG * DD * 4;

    const int TB = 256;
    const int gND  = NN * DD / TB;          // 16384 blocks
    const int gE   = NETOT / 8;             // 69632 blocks (wave per edge)
    const int gE4  = NETOT * NHEADS / TB;   // 8704 blocks
    const int gGD  = GG * DD / TB;          // 16 blocks

    for (int g = 0; g < 2; ++g) {
        const int* srcA = ei[g];
        const int* dstA = ei[g] + EE;

        k_convert_bf16<<<gND, TB, 0, stream>>>(x[g], XB);

        for (int L = 0; L < 3; ++L) {
            int pb = 2 + L * 8;
            const float* Wl    = (const float*)d_in[pb + 0];
            const float* bl    = (const float*)d_in[pb + 1];
            const float* Wr    = (const float*)d_in[pb + 2];
            const float* br    = (const float*)d_in[pb + 3];
            const float* att   = (const float*)d_in[pb + 4];
            const float* bias  = (const float*)d_in[pb + 5];
            const float* gamma = (const float*)d_in[pb + 6];
            const float* beta  = (const float*)d_in[pb + 7];

            k_swizzleW<<<2 * DD * DD / TB, TB, 0, stream>>>(Wl, Wr, WSWZ);
            k_wmma_gemm<<<NN / 128, TB, 0, stream>>>(XB, WSWZ, bl, XL, NN);
            k_wmma_gemm<<<NN / 128, TB, 0, stream>>>(XB, WSWZ + DD * DD, br, XR, NN);
            k_init_layer<<<gND, TB, 0, stream>>>(OUTB, bias, MEN, DEN, BNS, BNQ);
            k_edge1<<<gE, TB, 0, stream>>>(XL, XR, srcA, dstA, att, LG, MEN);
            k_edge2<<<gE4, TB, 0, stream>>>(dstA, LG, MEN, DEN);
            k_edge3<<<gE, TB, 0, stream>>>(XL, srcA, dstA, LG, DEN, OUTB);
            k_bn_stats<<<256, DD, 0, stream>>>(OUTB, BNS, BNQ);
            k_bn_fin<<<1, DD, 0, stream>>>(BNS, BNQ, gamma, beta, BSC, BSH);
            k_bn_apply<<<gND, TB, 0, stream>>>(OUTB, BSC, BSH, XF, XB);
        }

        k_pool_init<<<gGD, TB, 0, stream>>>(PS, PM, PC);
        k_pool_acc<<<gND, TB, 0, stream>>>(XF, bt[g], PS, PM, PC);
        k_pool_fin<<<gGD, TB, 0, stream>>>(PS, PM, PC, GB[g]);
    }

    k_head<<<1, TB, 0, stream>>>(GB[0], GB[1],
                                 (const float*)d_in[30], (const float*)d_in[31],   // Wv bv
                                 (const float*)d_in[32], (const float*)d_in[33],   // Wo bo
                                 (const float*)d_in[34], (const float*)d_in[35],   // W1 b1
                                 (const float*)d_in[36], (const float*)d_in[37],   // W2 b2
                                 (const float*)d_in[38], (const float*)d_in[39],   // W3 b3
                                 (float*)d_out);
}
